// CRNet_84653805404721
// MI455X (gfx1250) — compile-verified, matrix operand order checked by reading, not run on hardware
//
#include <hip/hip_runtime.h>
#include <hip/hip_bf16.h>

// ---------------------------------------------------------------------------
// CDNA5 (gfx1250) wave32 WMMA types
// ---------------------------------------------------------------------------
typedef __attribute__((ext_vector_type(16))) __bf16 v16bf;
typedef __attribute__((ext_vector_type(8)))  float  v8f;

// Problem constants
#define BB 1024
#define CC 256
#define KK 100
#define KKP 112          // cluster rows padded to multiple of 16
#define VDD 64
#define SDD 200
#define SDDP 224         // semantic K padded to multiple of 32
#define DD 256           // every GEMM here has N == 256
#define EE 6

// ---------------------------------------------------------------------------
// Packed-B layout (fragment-native bf16):
//   Wp[ kt*8192 + col*32 + h*16 + j ]  =  W[ kt*32 + h*8 + j + (j>=8 ? 8 : 0), col ]
// so a lane (h = lane>>4, col = n0+(lane&15)) reads its whole 32-K fragment as
// one contiguous 32-byte v16bf load. Matches ISA 7.12.2 B layout for 16x16x32.
// ---------------------------------------------------------------------------
__global__ __launch_bounds__(256)
void pack_w_bf16(const float* __restrict__ W, __bf16* __restrict__ Wp,
                 int Kp, int Ksrc) {
    int idx = blockIdx.x * 256 + threadIdx.x;
    if (idx >= Kp * DD) return;
    int j  = idx & 15;
    int h  = (idx >> 4) & 1;
    int c  = (idx >> 5) & 255;
    int kt = idx >> 13;
    int k  = kt * 32 + h * 8 + j + ((j >= 8) ? 8 : 0);
    float v = (k < Ksrc) ? W[(size_t)k * DD + c] : 0.0f;
    Wp[idx] = (__bf16)v;
}

// Zero-padded 2D copy: dst(Mp x Kp) <- src(Ms x Ks)
__global__ __launch_bounds__(256)
void pad_copy(const float* __restrict__ src, float* __restrict__ dst,
              int Ms, int Ks, int Mp, int Kp) {
    int idx = blockIdx.x * 256 + threadIdx.x;
    if (idx >= Mp * Kp) return;
    int r = idx / Kp, c = idx - r * Kp;
    dst[idx] = (r < Ms && c < Ks) ? src[(size_t)r * Ks + c] : 0.0f;
}

// A-fragment: 16 f32 -> bf16, four unconditional float4 loads (16B aligned).
__device__ __forceinline__ v16bf a_frag_cvt(const float* __restrict__ p) {
    float4 x0 = *(const float4*)(p);
    float4 x1 = *(const float4*)(p + 4);
    float4 x2 = *(const float4*)(p + 16);
    float4 x3 = *(const float4*)(p + 20);
    v16bf f;
    f[0]=(__bf16)x0.x;  f[1]=(__bf16)x0.y;  f[2]=(__bf16)x0.z;  f[3]=(__bf16)x0.w;
    f[4]=(__bf16)x1.x;  f[5]=(__bf16)x1.y;  f[6]=(__bf16)x1.z;  f[7]=(__bf16)x1.w;
    f[8]=(__bf16)x2.x;  f[9]=(__bf16)x2.y;  f[10]=(__bf16)x2.z; f[11]=(__bf16)x2.w;
    f[12]=(__bf16)x3.x; f[13]=(__bf16)x3.y; f[14]=(__bf16)x3.z; f[15]=(__bf16)x3.w;
    return f;
}

__device__ __forceinline__ v16bf a_frag_diff(const float* __restrict__ s,
                                             const float* __restrict__ c) {
    float4 s0 = *(const float4*)(s);      float4 c0 = *(const float4*)(c);
    float4 s1 = *(const float4*)(s + 4);  float4 c1 = *(const float4*)(c + 4);
    float4 s2 = *(const float4*)(s + 16); float4 c2 = *(const float4*)(c + 16);
    float4 s3 = *(const float4*)(s + 20); float4 c3 = *(const float4*)(c + 20);
    v16bf f;
    f[0]=(__bf16)(s0.x-c0.x);  f[1]=(__bf16)(s0.y-c0.y);
    f[2]=(__bf16)(s0.z-c0.z);  f[3]=(__bf16)(s0.w-c0.w);
    f[4]=(__bf16)(s1.x-c1.x);  f[5]=(__bf16)(s1.y-c1.y);
    f[6]=(__bf16)(s1.z-c1.z);  f[7]=(__bf16)(s1.w-c1.w);
    f[8]=(__bf16)(s2.x-c2.x);  f[9]=(__bf16)(s2.y-c2.y);
    f[10]=(__bf16)(s2.z-c2.z); f[11]=(__bf16)(s2.w-c2.w);
    f[12]=(__bf16)(s3.x-c3.x); f[13]=(__bf16)(s3.y-c3.y);
    f[14]=(__bf16)(s3.z-c3.z); f[15]=(__bf16)(s3.w-c3.w);
    return f;
}

// ---------------------------------------------------------------------------
// GEMM: D(Mx256) = act(A(MxK) @ W(Kx256) + bias).  M%16==0, K%32==0.
// block(32,4): one wave per 16x16 tile. No predication anywhere in hot loop.
// ---------------------------------------------------------------------------
__global__ __launch_bounds__(128)
void gemm_bf16_packed(const float* __restrict__ A, const __bf16* __restrict__ Wp,
                      const float* __restrict__ bias, float* __restrict__ D,
                      int K, int act) {
    int lane = threadIdx.x;
    int n0 = (blockIdx.x * 4 + threadIdx.y) * 16;
    int m0 = blockIdx.y * 16;
    int col = n0 + (lane & 15);
    int h   = (lane >> 4) & 1;

    const float*  Ap = A  + (size_t)(m0 + (lane & 15)) * K + h * 8;
    const __bf16* Bp = Wp + (size_t)col * 32 + h * 16;

    v8f acc = {};
    int nt = K >> 5;
    for (int kt = 0; kt < nt; ++kt) {
        v16bf a = a_frag_cvt(Ap + kt * 32);
        v16bf b = *(const v16bf*)(Bp + (size_t)kt * 8192);
        acc = __builtin_amdgcn_wmma_f32_16x16x32_bf16(false, a, false, b,
                                                      (short)0, acc, false, false);
    }

    float bv = (bias != nullptr) ? bias[col] : 0.0f;
    int rbase = m0 + h * 8;
#pragma unroll
    for (int r = 0; r < 8; ++r) {
        float v = acc[r] + bv;
        if (act) v = v > 0.0f ? v : 0.0f;
        D[(size_t)(rbase + r) * DD + col] = v;
    }
}

// ---------------------------------------------------------------------------
// Fused expert einsum + relu + sum over (k,e):
//   S[p,o] = sum_{k,e} relu( sum_d (sem0[p,d]-ccom[k,d]) * Wexp[e,d,o] + bexp[e,o] )
// One wave per 16x16 tile of S; A-diff fragments hoisted across the 6 experts;
// f32 running sum stays in 8 VGPRs over all 600 inner 16x16x256 GEMMs.
// Wexpp is the packed (1536 x 256) view of W_exp: expert e = tiles e*8 .. e*8+7.
// ---------------------------------------------------------------------------
__global__ __launch_bounds__(128)
void expert_sum_kernel(const float* __restrict__ sem0, const float* __restrict__ ccom,
                       const __bf16* __restrict__ Wexpp, const float* __restrict__ bexp,
                       float* __restrict__ S) {
    int lane = threadIdx.x;
    int n0 = (blockIdx.x * 4 + threadIdx.y) * 16;
    int m0 = blockIdx.y * 16;
    int col = n0 + (lane & 15);
    int h   = (lane >> 4) & 1;

    const float*  semrow = sem0  + (size_t)(m0 + (lane & 15)) * DD + h * 8;
    const __bf16* Bp     = Wexpp + (size_t)col * 32 + h * 16;

    float be[EE];
#pragma unroll
    for (int e = 0; e < EE; ++e) be[e] = bexp[e * DD + col];

    v8f acc = {};
    for (int kk = 0; kk < KK; ++kk) {
        const float* clusrow = ccom + (size_t)kk * DD + h * 8;
        v16bf a[8];
#pragma unroll
        for (int t = 0; t < 8; ++t)
            a[t] = a_frag_diff(semrow + t * 32, clusrow + t * 32);

#pragma unroll
        for (int e = 0; e < EE; ++e) {
            v8f c = {};
#pragma unroll
            for (int t = 0; t < 8; ++t) {
                v16bf b = *(const v16bf*)(Bp + (size_t)(e * 8 + t) * 8192);
                c = __builtin_amdgcn_wmma_f32_16x16x32_bf16(false, a[t], false, b,
                                                            (short)0, c, false, false);
            }
#pragma unroll
            for (int r = 0; r < 8; ++r) {
                float v = c[r] + be[e];
                acc[r] += (v > 0.0f) ? v : 0.0f;
            }
        }
    }

    int rbase = m0 + h * 8;
#pragma unroll
    for (int r = 0; r < 8; ++r)
        S[(size_t)(rbase + r) * DD + col] = acc[r];
}

// ---------------------------------------------------------------------------
// Fused relation head:
//   relations[b,c] = sum_d relu(va[b,d] + sb[c,d] + br1[d]) * w2[d] + br2
// LDS-tiled; the 268 MB f32 h tensor is never materialized.
// ---------------------------------------------------------------------------
__global__ __launch_bounds__(256)
void relation_kernel(const float* __restrict__ va, const float* __restrict__ sbb,
                     const float* __restrict__ br1, const float* __restrict__ w2,
                     const float* __restrict__ br2, float* __restrict__ out) {
    __shared__ float vas[16][DD];      // broadcast row access -> conflict-free
    __shared__ float sbs[32][DD + 1];  // padded: per-lane row access
    __shared__ float w2s[DD];

    int tid = threadIdx.x;
    int b0 = blockIdx.y * 16;
    int c0 = blockIdx.x * 32;

    for (int i = tid; i < 16 * DD; i += 256) {
        int r = i >> 8, d = i & (DD - 1);
        vas[r][d] = va[(size_t)(b0 + r) * DD + d];
    }
    for (int i = tid; i < 32 * DD; i += 256) {
        int r = i >> 8, d = i & (DD - 1);
        sbs[r][d] = sbb[(size_t)(c0 + r) * DD + d] + br1[d];  // fold b_r1
    }
    w2s[tid & (DD - 1)] = w2[tid & (DD - 1)];
    __syncthreads();

    int tx = tid & 31;      // c within tile
    int ty = tid >> 5;      // 0..7 -> b rows ty and ty+8
    float a0 = 0.0f, a1 = 0.0f;
#pragma unroll 4
    for (int d = 0; d < DD; ++d) {
        float wv = w2s[d];
        float s  = sbs[tx][d];
        float h0 = vas[ty][d] + s;
        float h1 = vas[ty + 8][d] + s;
        a0 += (h0 > 0.0f ? h0 : 0.0f) * wv;
        a1 += (h1 > 0.0f ? h1 : 0.0f) * wv;
    }
    float b2 = br2[0];
    out[(size_t)(b0 + ty)     * CC + (c0 + tx)] = a0 + b2;
    out[(size_t)(b0 + ty + 8) * CC + (c0 + tx)] = a1 + b2;
}

// ---------------------------------------------------------------------------
// Host-side orchestration (graph-capture safe: only kernel launches on stream)
// ---------------------------------------------------------------------------
static inline void launch_gemm(const float* A, const __bf16* Wp, const float* bias,
                               float* D, int M, int K, int act, hipStream_t s) {
    dim3 grid(4, M / 16);           // N==256 -> 16 tiles -> grid.x = 4 exact
    dim3 block(32, 4);
    gemm_bf16_packed<<<grid, block, 0, s>>>(A, Wp, bias, D, K, act);
}

static inline void launch_pack(const float* W, __bf16* Wp, int Kp, int Ksrc,
                               hipStream_t s) {
    pack_w_bf16<<<(Kp * DD + 255) / 256, 256, 0, s>>>(W, Wp, Kp, Ksrc);
}

extern "C" void kernel_launch(void* const* d_in, const int* in_sizes, int n_in,
                              void* d_out, int out_size, void* d_ws, size_t ws_size,
                              hipStream_t stream) {
    const float* vf   = (const float*)d_in[0];   // (1024,64)
    const float* cp   = (const float*)d_in[1];   // (256,200)
    const float* cc   = (const float*)d_in[2];   // (100,64)
    const float* Wv1  = (const float*)d_in[3];   // (64,256)
    const float* bv1  = (const float*)d_in[4];
    const float* Wv2  = (const float*)d_in[5];   // (256,256)
    const float* bv2  = (const float*)d_in[6];
    const float* Ws1  = (const float*)d_in[7];   // (200,256)
    const float* bs1  = (const float*)d_in[8];
    const float* Ws2  = (const float*)d_in[9];   // (256,256)
    const float* bs2  = (const float*)d_in[10];
    const float* Wexp = (const float*)d_in[11];  // (6,256,256) == (1536,256)
    const float* bexp = (const float*)d_in[12];  // (6,256)
    const float* Wr1  = (const float*)d_in[13];  // (512,256)
    const float* br1  = (const float*)d_in[14];  // (256)
    const float* Wr2  = (const float*)d_in[15];  // (256,1)
    const float* br2  = (const float*)d_in[16];  // (1)
    float* out = (float*)d_out;                  // (1024,256)

    // ---- workspace layout: f32 scratch, then 32B-aligned bf16 packed weights
    float* ws = (float*)d_ws;
    float* t1   = ws;                   // 1024*256
    float* vcom = t1   + BB * DD;       // 1024*256
    float* ccp  = vcom + BB * DD;       // 112*64   (padded cluster_centers)
    float* t2   = ccp  + KKP * VDD;     // 112*256
    float* ccom = t2   + KKP * DD;      // 112*256  (rows >=100 unused)
    float* cpp  = ccom + KKP * DD;      // 256*224  (padded class_prototypes)
    float* t3   = cpp  + CC * SDDP;     // 256*256
    float* sem0 = t3   + CC * DD;       // 256*256
    float* S    = sem0 + CC * DD;       // 256*256
    float* va   = S    + CC * DD;       // 1024*256
    float* sbb  = va   + BB * DD;       // 256*256
    float* fend = sbb  + CC * DD;

    __bf16* bw = (__bf16*)((((uintptr_t)fend) + 31) & ~(uintptr_t)31);
    __bf16* Wv1p  = bw;                 // 64*256
    __bf16* Wv2p  = Wv1p + VDD * DD;    // 256*256
    __bf16* Ws1p  = Wv2p + DD * DD;     // 224*256
    __bf16* Ws2p  = Ws1p + SDDP * DD;   // 256*256
    __bf16* Wexpp = Ws2p + DD * DD;     // 1536*256
    __bf16* Wr1p  = Wexpp + EE * DD * DD; // 512*256

    // ---- pre-pack weights (bf16, fragment-native) and pad ragged inputs
    launch_pack(Wv1, Wv1p, VDD, VDD, stream);
    launch_pack(Wv2, Wv2p, DD, DD, stream);
    launch_pack(Ws1, Ws1p, SDDP, SDD, stream);          // zero-pad K 200->224
    launch_pack(Ws2, Ws2p, DD, DD, stream);
    launch_pack(Wexp, Wexpp, EE * DD, EE * DD, stream); // 6 experts as 1536 rows
    launch_pack(Wr1, Wr1p, 2 * DD, 2 * DD, stream);     // Wa tiles 0-7, Wb 8-15
    pad_copy<<<(KKP * VDD + 255) / 256, 256, 0, stream>>>(cc, ccp, KK, VDD, KKP, VDD);
    pad_copy<<<(CC * SDDP + 255) / 256, 256, 0, stream>>>(cp, cpp, CC, SDD, CC, SDDP);

    // ---- MLPs (all dims padded: no predication in GEMM hot loops)
    launch_gemm(vf,  Wv1p, bv1, t1,   BB,  VDD,  1, stream);
    launch_gemm(t1,  Wv2p, bv2, vcom, BB,  DD,   0, stream);
    launch_gemm(ccp, Wv1p, bv1, t2,   KKP, VDD,  1, stream);
    launch_gemm(t2,  Wv2p, bv2, ccom, KKP, DD,   0, stream);
    launch_gemm(cpp, Ws1p, bs1, t3,   CC,  SDDP, 1, stream);
    launch_gemm(t3,  Ws2p, bs2, sem0, CC,  DD,   0, stream);

    // ---- fused expert einsum (~20.1 GFLOP of the ~21 GF total)
    {
        dim3 grid(4, 16);
        dim3 block(32, 4);
        expert_sum_kernel<<<grid, block, 0, stream>>>(sem0, ccom, Wexpp, bexp, S);
    }

    // ---- relation projections: Wa = Wr1 tiles 0..7, Wb = tiles 8..15
    launch_gemm(vcom, Wr1p,                 nullptr, va,  BB, DD, 0, stream);
    launch_gemm(S,    Wr1p + (size_t)DD * 32 * 8, nullptr, sbb, CC, DD, 0, stream);

    // ---- fused broadcast-add + relu + dot(W_r2) + b_r2
    {
        dim3 grid(CC / 32, BB / 16);
        dim3 block(256);
        relation_kernel<<<grid, block, 0, stream>>>(va, sbb, br1, Wr2, br2, out);
    }
}
